// IdnRNN_30554397344493
// MI455X (gfx1250) — matrix operand order repaired and checked by reading, not compiled, not run on hardware
//
#include <hip/hip_runtime.h>
#include <hip/hip_bf16.h>

// Problem dims (fixed by the reference)
#define B_ 64
#define T_ 512
#define I_ 512
#define H_ 1024
#define SCAN_WGS 32            // persistent grid for the recurrence
#define BTH_ ((size_t)B_ * T_ * H_)

typedef __bf16 bf16;
typedef __attribute__((ext_vector_type(16))) bf16  v16bf;
typedef __attribute__((ext_vector_type(8)))  bf16  v8bf;
typedef __attribute__((ext_vector_type(8)))  float v8f;

typedef __attribute__((ext_vector_type(4))) unsigned int u32x4;
typedef __attribute__((ext_vector_type(8))) int          i32x8;
typedef __attribute__((ext_vector_type(4))) int          i32x4;

// ---------------------------------------------------------------------------
// WMMA helper: D(f32 16x16) = A(bf16 16x32) * B(bf16 32x16) + C
// ---------------------------------------------------------------------------
__device__ __forceinline__ v8f wmma_bf16(v16bf a, v16bf b, v8f c) {
  return __builtin_amdgcn_wmma_f32_16x16x32_bf16(
      /*neg_a=*/false, a, /*neg_b=*/false, b,
      /*c_mod=*/(short)0, c, /*reuse_a=*/false, /*reuse_b=*/false);
}

// ---------------------------------------------------------------------------
// TDM: 1-D contiguous load of n8*8 bytes from global -> LDS offset 0.
// D# per CDNA5 ISA sec 8.3/8.4: group0 {count=1, lds_addr, global_addr, type=2},
// group1 {data_size=3 (8B), tensor_dim0=tile_dim0=n8, dim1=1, stride0=n8},
// groups 2/3 zero (tile_dim2..4 = 0 = unused).
// clang-23 toolchain: 6-arg builtin (g0, g1, g2, g3, g4, cpol).
// ---------------------------------------------------------------------------
__device__ __forceinline__ void tdm_load_1d_to_lds(const void* gsrc,
                                                   unsigned lds_byte_off,
                                                   unsigned n8 /* 8-byte elems, <= 65535 */) {
  unsigned long long ga = (unsigned long long)gsrc;
  u32x4 g0;
  g0[0] = 1u;                                          // count=1 (valid), user mode
  g0[1] = lds_byte_off;                                // lds_addr (bytes)
  g0[2] = (unsigned)(ga & 0xFFFFFFFFu);                // global_addr[31:0]
  g0[3] = ((unsigned)(ga >> 32) & 0x01FFFFFFu) | (2u << 30); // addr[56:32] | type=2

  i32x8 g1;
  g1[0] = (int)(3u << 16);                             // data_size=3 (8 bytes)
  g1[1] = (int)((n8 & 0xFFFFu) << 16);                 // tensor_dim0[15:0] @ bits 63:48
  g1[2] = (int)(((n8 >> 16) & 0xFFFFu) | (1u << 16));  // tensor_dim0[31:16], tensor_dim1=1
  g1[3] = (int)((n8 & 0xFFFFu) << 16);                 // tensor_dim1[31:16]=0, tile_dim0=n8
  g1[4] = 0;                                           // tile_dim1=0, tile_dim2=0 (unused)
  g1[5] = (int)n8;                                     // tensor_dim0_stride[31:0]
  g1[6] = 0;                                           // stride0[47:32], dim1_stride[15:0]
  g1[7] = 0;

  i32x4 gz4 = {0, 0, 0, 0};
  i32x8 gz8 = {0, 0, 0, 0, 0, 0, 0, 0};
  __builtin_amdgcn_tensor_load_to_lds(g0, g1, gz4, gz4, gz8, /*cpol=*/0);
}

// ---------------------------------------------------------------------------
// Pack Wh (f32 [H][H], h_new[j] = sum_k h[k]*Wh[j][k]) into per-lane bf16
// B-fragment layout:
//   whp[(((ntile*32 + kc)*32 + lane)*16 + j] = Wh[col][k]
//   col = ntile*16 + (lane&15),  k = kc*32 + (lane>>4)*16 + j
// ---------------------------------------------------------------------------
__global__ void pack_wh_kernel(const float* __restrict__ Wh,
                               bf16* __restrict__ whp) {
  int u = blockIdx.x * blockDim.x + threadIdx.x;    // 65536 threads
  int L  = u & 31;
  int kc = (u >> 5) & 31;
  int c  = u >> 10;
  int col   = c * 16 + (L & 15);
  int kbase = kc * 32 + (L >> 4) * 16;
  const float* src = Wh + (size_t)col * H_ + kbase; // contiguous in j
  v16bf v;
#pragma unroll
  for (int j = 0; j < 16; ++j) v[j] = (bf16)src[j];
  *(v16bf*)(whp + (size_t)u * 16) = v;
}

// Same packing for Wi (f32 [H][I]): B[i][h] = Wi[h][i]
__global__ void pack_wi_kernel(const float* __restrict__ Wi,
                               bf16* __restrict__ wip) {
  int u = blockIdx.x * blockDim.x + threadIdx.x;    // 32768 threads
  int L  = u & 31;
  int kc = (u >> 5) & 15;                           // I/32 = 16 k-chunks
  int c  = u >> 9;
  int col   = c * 16 + (L & 15);
  int kbase = kc * 32 + (L >> 4) * 16;
  const float* src = Wi + (size_t)col * I_ + kbase;
  v16bf v;
#pragma unroll
  for (int j = 0; j < 16; ++j) v[j] = (bf16)src[j];
  *(v16bf*)(wip + (size_t)u * 16) = v;
}

// ---------------------------------------------------------------------------
// Phase 1: pre[b*T+t][h] = x[b][t][:] . Wi[h][:] + bi[h] + bh[h]
// Written straight into d_out (the scan overwrites it in place).
// ---------------------------------------------------------------------------
__global__ void gemm_pre_kernel(const float* __restrict__ x,
                                const bf16*  __restrict__ wip,
                                const float* __restrict__ bi,
                                const float* __restrict__ bh,
                                float* __restrict__ out) {
  const int lane = threadIdx.x & 31;
  const int w    = threadIdx.x >> 5;
  const int gw   = blockIdx.x * 8 + w;
  const int ntile = gw & 63;
  const int mtile = gw >> 6;               // 0..2047
  const int lr = lane & 15;
  const int lh = lane >> 4;

  const int colg = ntile * 16 + lr;        // C/D column for this lane
  const int arow = mtile * 16 + lr;        // A row for this lane

  v8f acc;
  const float bias = bi[colg] + bh[colg];
#pragma unroll
  for (int r = 0; r < 8; ++r) acc[r] = bias;

  const float* xrow = x + (size_t)arow * I_;
#pragma unroll 4
  for (int kc = 0; kc < I_ / 32; ++kc) {
    const float* ap = xrow + kc * 32 + lh * 16;
    v8f lo = *(const v8f*)(ap);            // k = kc*32 + lh*16 + 0..7
    v8f hi = *(const v8f*)(ap + 8);        // k = kc*32 + lh*16 + 8..15
    v16bf a;
#pragma unroll
    for (int j = 0; j < 8; ++j) { a[j] = (bf16)lo[j]; a[8 + j] = (bf16)hi[j]; }
    v16bf b = *(const v16bf*)(wip + (((size_t)ntile * (I_ / 32) + kc) * 32 + lane) * 16);
    acc = wmma_bf16(a, b, acc);
  }

  const size_t obase = (size_t)(mtile * 16 + lh * 8) * H_ + colg;
#pragma unroll
  for (int r = 0; r < 8; ++r) out[obase + (size_t)r * H_] = acc[r];
}

// ---------------------------------------------------------------------------
// Cumulative grid barrier (counter memset to 0 per launch).
// ---------------------------------------------------------------------------
__device__ __forceinline__ void grid_barrier(unsigned* cnt, unsigned target) {
  __syncthreads();
  if (threadIdx.x == 0) {
    __threadfence();                                        // release stores
    __hip_atomic_fetch_add(cnt, 1u, __ATOMIC_RELEASE, __HIP_MEMORY_SCOPE_AGENT);
    while (__hip_atomic_load(cnt, __ATOMIC_ACQUIRE, __HIP_MEMORY_SCOPE_AGENT) < target)
      __builtin_amdgcn_s_sleep(2);
  }
  __syncthreads();
  __threadfence();                                          // acquire: drop stale L0 lines
}

// ---------------------------------------------------------------------------
// Phase 2: persistent scan. 32 WGs x 8 waves = 256 waves = 256 tiles/step.
//   h_t = relu(pre_t + h_{t-1} @ Wh^T)
// Block bid owns ntiles [4*(bid&15) .. +3] x brows [2*(bid>>4) .. +1].
// Its 4 ntile B-slices (128 KB contiguous in whp) are TDM-loaded into LDS
// once; all 512 steps read B fragments from LDS (ds_load_b128).
// ---------------------------------------------------------------------------
__global__ void scan_kernel(float* __restrict__ out,
                            const bf16* __restrict__ whp,
                            bf16* __restrict__ hbuf,      // 2 * B_*H_ bf16
                            unsigned* __restrict__ cnt) {
  extern __shared__ char lds_raw[];                       // 128 KB dynamic LDS
  const bf16* blds = (const bf16*)lds_raw;

  const int lane = threadIdx.x & 31;
  const int w    = threadIdx.x >> 5;
  const int ngrp = blockIdx.x & 15;
  const int bgrp = blockIdx.x >> 4;
  const int nslot = w & 3;                 // ntile slot within LDS
  const int ntile = ngrp * 4 + nslot;      // 0..63 (H tiles)
  const int brow  = bgrp * 2 + (w >> 2);   // 0..3  (B tiles)
  const int lr = lane & 15;
  const int lh = lane >> 4;

  const int colg  = ntile * 16 + lr;
  const int arow  = brow * 16 + lr;        // row of h_{t-1} this lane streams
  const int crow0 = brow * 16 + lh * 8;    // first C/D row for this lane

  // --- Stage this block's Wh fragments into LDS with the Tensor Data Mover.
  // whp slice: 4 ntiles * 32 kc * 32 lanes * 16 bf16 = 128 KB contiguous.
  if (w == 0) {
    const bf16* src = whp + (size_t)ngrp * 4 * 32 * 32 * 16;
    tdm_load_1d_to_lds(src, 0, /*n8=*/(4 * 32 * 32 * 16 * 2) / 8);  // 16384 x 8B
    __builtin_amdgcn_s_wait_tensorcnt(0);
  }
  __syncthreads();

  for (int t = 0; t < T_; ++t) {
    const bf16* hprev = hbuf + (size_t)(t & 1) * (B_ * H_);
    bf16*       hnext = hbuf + (size_t)((t + 1) & 1) * (B_ * H_);

    // acc <- pre_t tile (already includes bi+bh)
    const size_t obase = ((size_t)crow0 * T_ + t) * H_ + colg;
    v8f acc;
#pragma unroll
    for (int r = 0; r < 8; ++r) acc[r] = out[obase + (size_t)r * T_ * H_];
    if (t + 1 < T_) __builtin_prefetch(&out[obase + H_], 0, 3);  // next step's pre

    // K loop over H=1024 in chunks of 32; B fragments from LDS
    const bf16* hrow = hprev + (size_t)arow * H_;
#pragma unroll 4
    for (int kc = 0; kc < H_ / 32; ++kc) {
      const v8bf* pa = (const v8bf*)(hrow + kc * 32 + lh * 16);
      v8bf alo = pa[0];                    // k = kc*32 + lh*16 + 0..7
      v8bf ahi = pa[1];                    // k = kc*32 + lh*16 + 8..15
      v16bf a = __builtin_shufflevector(alo, ahi,
                                        0, 1, 2, 3, 4, 5, 6, 7,
                                        8, 9, 10, 11, 12, 13, 14, 15);
      v16bf b = *(const v16bf*)(blds + (((size_t)nslot * 32 + kc) * 32 + lane) * 16);
      acc = wmma_bf16(a, b, acc);
    }

    // relu
#pragma unroll
    for (int r = 0; r < 8; ++r) acc[r] = fmaxf(acc[r], 0.0f);

    // write h_t: f32 into d_out (= outs[b][t][:]), bf16 into next buffer
#pragma unroll
    for (int r = 0; r < 8; ++r) {
      out[obase + (size_t)r * T_ * H_] = acc[r];
      hnext[(size_t)(crow0 + r) * H_ + colg] = (bf16)acc[r];
    }
    if (t == T_ - 1) {                     // final hidden state
#pragma unroll
      for (int r = 0; r < 8; ++r)
        out[BTH_ + (size_t)(crow0 + r) * H_ + colg] = acc[r];
    }

    grid_barrier(cnt, (unsigned)(t + 1) * SCAN_WGS);
  }
}

// ---------------------------------------------------------------------------
// Launch: pack weights -> pre-GEMM -> persistent scan (all on `stream`).
// Workspace layout (recomputed every call; deterministic):
//   [0, 2MB)          whp  (Wh packed bf16 fragments)
//   [2MB, 3MB)        wip  (Wi packed bf16 fragments)
//   [3MB, 3MB+256KB)  hbuf (double-buffered bf16 h; buf0 zeroed = h0)
//   [+512KB, +4B)     barrier counter
// ---------------------------------------------------------------------------
extern "C" void kernel_launch(void* const* d_in, const int* in_sizes, int n_in,
                              void* d_out, int out_size, void* d_ws, size_t ws_size,
                              hipStream_t stream) {
  const float* x  = (const float*)d_in[0];
  const float* Wi = (const float*)d_in[1];
  const float* bi = (const float*)d_in[2];
  const float* Wh = (const float*)d_in[3];
  const float* bh = (const float*)d_in[4];
  float* out = (float*)d_out;

  char* ws = (char*)d_ws;
  bf16*     whp  = (bf16*)(ws);
  bf16*     wip  = (bf16*)(ws + (2u << 20));
  bf16*     hbuf = (bf16*)(ws + (3u << 20));
  unsigned* cnt  = (unsigned*)(ws + (3u << 20) + (512u << 10));

  // h0 = 0 (first step's A-matrix) and reset barrier counter
  (void)hipMemsetAsync(hbuf, 0, (size_t)B_ * H_ * sizeof(bf16), stream);
  (void)hipMemsetAsync(cnt, 0, sizeof(unsigned), stream);

  pack_wh_kernel<<<256, 256, 0, stream>>>(Wh, whp);   // 65536 threads
  pack_wi_kernel<<<128, 256, 0, stream>>>(Wi, wip);   // 32768 threads

  // 131072 tiles, 8 waves/block -> 16384 blocks
  gemm_pre_kernel<<<16384, 256, 0, stream>>>(x, wip, bi, bh, out);

  // 128 KB dynamic LDS per block for the TDM-staged Wh fragments
  scan_kernel<<<SCAN_WGS, 256, 128 * 1024, stream>>>(out, whp, hbuf, cnt);
}
